// ShidoTreeLSTMLayer_31439160606821
// MI455X (gfx1250) — matrix operand order repaired and checked by reading, not compiled, not run on hardware
//
#include <hip/hip_runtime.h>

typedef __bf16 bf16;
typedef bf16  v16bf __attribute__((ext_vector_type(16)));
typedef float v8f   __attribute__((ext_vector_type(8)));
typedef unsigned int u32;
typedef u32 u32x4 __attribute__((ext_vector_type(4)));

#define LVLS 8
#define NN   8192
#define KK   16
#define DIN  256
#define HD   128
#define G4   512

union FragA { v16bf v; u32x4 q[2]; };

__device__ __forceinline__ v8f splat8(float x) {
  v8f r;
#pragma unroll
  for (int i = 0; i < 8; ++i) r[i] = x;
  return r;
}

__device__ __forceinline__ v16bf zero_a() {
  FragA f; u32x4 z = {0u, 0u, 0u, 0u};
  f.q[0] = z; f.q[1] = z;
  return f.v;
}

// A-fragment (16x32 bf16): lane = M row; elems 0..7 -> K = kf*32 + lhalf*8 + i,
// elems 8..15 -> K = kf*32 + 16 + lhalf*8 + i   (ISA 7.12.2)
__device__ __forceinline__ v16bf load_a_bf16(const bf16* row, int kf, int lhalf) {
  FragA f;
  const bf16* p = row + kf * 32 + lhalf * 8;
  f.q[0] = *(const u32x4*)(p);
  f.q[1] = *(const u32x4*)(p + 16);
  return f.v;
}

__device__ __forceinline__ v16bf load_a_f32(const float* row, int kf, int lhalf) {
  v16bf v;
  const float* p0 = row + kf * 32 + lhalf * 8;
  const float* p1 = p0 + 16;
#pragma unroll
  for (int i = 0; i < 8; ++i) { v[i] = (bf16)p0[i]; v[8 + i] = (bf16)p1[i]; }
  return v;
}

// B-fragment (32x16 bf16): lane = col (lrow); elems 0..7 -> K = kbase + lhalf*8 + i,
// elems 8..15 -> K = kbase + 16 + lhalf*8 + i
// mat is row-major [cols][stride], B[k][col] = mat[col][k]
__device__ __forceinline__ v16bf load_b_bf16(const bf16* mat, int col, int stride,
                                             int kbase, int lhalf) {
  FragA f;
  const bf16* p = mat + (size_t)col * stride + kbase + lhalf * 8;
  f.q[0] = *(const u32x4*)(p);
  f.q[1] = *(const u32x4*)(p + 16);
  return f.v;
}

__device__ __forceinline__ v8f wmma_bf16(v16bf a, v16bf b, v8f c) {
  return __builtin_amdgcn_wmma_f32_16x16x32_bf16(false, a, false, b, (short)0, c,
                                                 false, false);
}

__device__ __forceinline__ float sigm(float x) { return 1.0f / (1.0f + __expf(-x)); }

// CDNA5 async global->LDS copy, 16B per lane, tracked by ASYNCcnt.
__device__ __forceinline__ void async_copy_b128(u32 lds_off, const void* gptr) {
  asm volatile("global_load_async_to_lds_b128 %0, %1, off"
               :: "v"(lds_off), "v"(gptr) : "memory");
}
__device__ __forceinline__ void wait_async0() {
  asm volatile("s_wait_asynccnt 0" ::: "memory");
}

// ---------------- prep kernels ----------------
__global__ __launch_bounds__(256) void cvt_kernel(const float* __restrict__ src,
                                                  bf16* __restrict__ dst, int n) {
  int i = blockIdx.x * 256 + threadIdx.x;
  if (i < n) dst[i] = (bf16)src[i];
}

__global__ __launch_bounds__(256) void bias_kernel(const float* __restrict__ a,
                                                   const float* __restrict__ b,
                                                   float* __restrict__ o) {
  int i = blockIdx.x * 256 + threadIdx.x;
  if (i < G4) o[i] = a[i] + b[i];
}

// ---------------- per-level elementwise kernels ----------------
__global__ __launch_bounds__(256) void gather_kernel(const int* __restrict__ ind,
                                                     const float* __restrict__ htab,
                                                     bf16* __restrict__ hfw) {
  size_t id = (size_t)blockIdx.x * 256 + threadIdx.x;   // over N*K*H
  int n = (int)(id >> 11);
  int rem = (int)(id & 2047);
  int k = rem >> 7, j = rem & 127;
  int iv = ind[n * KK + k];
  float v = 0.0f;
  if (htab != nullptr && iv != -1) v = htab[(size_t)(iv - 1) * HD + j];
  hfw[id] = (bf16)v;
}

__global__ __launch_bounds__(256) void len_kernel(const int* __restrict__ ind,
                                                  int* __restrict__ lenp) {
  int n = blockIdx.x * 256 + threadIdx.x;
  if (n < NN) {
    int c = 0;
#pragma unroll
    for (int k = 0; k < KK; ++k) c += (ind[n * KK + k] != -1) ? 1 : 0;
    lenp[n] = c;
  }
}

// Wx = x @ W_w^T + W_b : [8192, 512], K = 256
__global__ __launch_bounds__(128) void wx_kernel(const float* __restrict__ xl,
                                                 const bf16* __restrict__ wwb,
                                                 const float* __restrict__ wb,
                                                 float* __restrict__ wx) {
  const int wave = threadIdx.x >> 5, lane = threadIdx.x & 31;
  const int lrow = lane & 15, lhalf = lane >> 4;
  const int nodebase = (blockIdx.x * 4 + wave) * 16;
  const float* arow = xl + (size_t)(nodebase + lrow) * DIN;
  v16bf a[8];
#pragma unroll
  for (int kf = 0; kf < 8; ++kf) a[kf] = load_a_f32(arow, kf, lhalf);
  for (int nt = 0; nt < 32; ++nt) {
    v8f acc = splat8(wb[nt * 16 + lrow]);
#pragma unroll
    for (int kf = 0; kf < 8; ++kf)
      acc = wmma_bf16(a[kf], load_b_bf16(wwb, nt * 16 + lrow, DIN, kf * 32, lhalf), acc);
#pragma unroll
    for (int r = 0; r < 8; ++r)
      wx[(size_t)(nodebase + r + 8 * lhalf) * G4 + nt * 16 + lrow] = acc[r];
  }
}

// ---------------- fused BiLSTM scan (one module, both directions) ----------------
__global__ __launch_bounds__(128)
void bilstm_kernel(const bf16* __restrict__ hfw, const int* __restrict__ lenp,
                   const bf16* __restrict__ WihF, const bf16* __restrict__ WhhF,
                   const bf16* __restrict__ WihB, const bf16* __restrict__ WhhB,
                   const float* __restrict__ biasF, const float* __restrict__ biasB,
                   const float* __restrict__ h0F, const float* __restrict__ c0F,
                   const float* __restrict__ h0B, const float* __restrict__ c0B,
                   bf16* __restrict__ yfseq, bf16* __restrict__ ybseq,
                   float* __restrict__ ylast, int storeSeq) {
  extern __shared__ char smem[];
  bf16* sWih = (bf16*)smem;                 // 512*128 bf16
  bf16* sWhh = sWih + G4 * HD;              // 512*128 bf16
  const int tid = threadIdx.x;
  const int wave = tid >> 5, lane = tid & 31;
  const int lrow = lane & 15, lhalf = lane >> 4;
  bf16* sHH = sWhh + G4 * HD + wave * (16 * HD);  // per-wave h_t transpose buffer
  const int nodebase = (blockIdx.x * 4 + wave) * 16;
  const int lnA = lenp[nodebase + lrow];
  int lastv[8];
#pragma unroll
  for (int r = 0; r < 8; ++r) {
    int ln = lenp[nodebase + r + 8 * lhalf];
    lastv[r] = (ln + 15) & 15;              // (len-1) mod 16, len==0 -> 15
  }
  const u32 ldsWih = (u32)(size_t)(void*)sWih;   // LDS byte address = addr[31:0]
  const u32 ldsWhh = (u32)(size_t)(void*)sWhh;
  for (int d = 0; d < 2; ++d) {
    const bf16* gWih = d ? WihB : WihF;
    const bf16* gWhh = d ? WhhB : WhhF;
    const float* bias = d ? biasB : biasF;
    const float* h0 = d ? h0B : h0F;
    const float* c0 = d ? c0B : c0F;
    bf16* yseq = d ? ybseq : yfseq;
    __syncthreads();   // all waves done reading previous direction's weights
    // async global->LDS staging of this direction's weights (128 KB each)
    {
      const char* gI = (const char*)gWih;
      const char* gH = (const char*)gWhh;
      for (int i = tid; i < (G4 * HD * 2) / 16; i += 128) {
        async_copy_b128(ldsWih + (u32)i * 16, gI + (size_t)i * 16);
        async_copy_b128(ldsWhh + (u32)i * 16, gH + (size_t)i * 16);
      }
      wait_async0();
    }
    __syncthreads();
    for (int idx = lane; idx < 16 * HD; idx += 32) sHH[idx] = (bf16)h0[idx & 127];
    v8f cc[8];
#pragma unroll
    for (int jt = 0; jt < 8; ++jt) cc[jt] = splat8(c0[jt * 16 + lrow]);
    for (int t = 0; t < KK; ++t) {
      v16bf ax[4], ah[4];
      int arow = d ? (lnA - 1 - t) : t;
      if (arow >= 0) {
        const bf16* row = hfw + ((size_t)(nodebase + lrow) * KK + arow) * HD;
#pragma unroll
        for (int kf = 0; kf < 4; ++kf) ax[kf] = load_a_bf16(row, kf, lhalf);
      } else {
#pragma unroll
        for (int kf = 0; kf < 4; ++kf) ax[kf] = zero_a();
      }
      // prefetch next step's A row while this step's WMMAs run
      {
        int nrow = d ? (lnA - 2 - t) : (t + 1);
        if (nrow >= 0 && nrow < KK)
          __builtin_prefetch(hfw + ((size_t)(nodebase + lrow) * KK + nrow) * HD, 0, 3);
      }
      {
        const bf16* row = sHH + lrow * HD;
#pragma unroll
        for (int kf = 0; kf < 4; ++kf) ah[kf] = load_a_bf16(row, kf, lhalf);
      }
#pragma unroll
      for (int jt = 0; jt < 8; ++jt) {
        v8f acc[4];
#pragma unroll
        for (int g = 0; g < 4; ++g) acc[g] = splat8(bias[g * 128 + jt * 16 + lrow]);
#pragma unroll
        for (int g = 0; g < 4; ++g) {
          const int colb = g * 128 + jt * 16 + lrow;
#pragma unroll
          for (int kf = 0; kf < 4; ++kf)
            acc[g] = wmma_bf16(ax[kf], load_b_bf16(sWih, colb, HD, kf * 32, lhalf), acc[g]);
#pragma unroll
          for (int kf = 0; kf < 4; ++kf)
            acc[g] = wmma_bf16(ah[kf], load_b_bf16(sWhh, colb, HD, kf * 32, lhalf), acc[g]);
        }
#pragma unroll
        for (int r = 0; r < 8; ++r) {
          float iN = sigm(acc[0][r]);
          float fN = sigm(acc[1][r]);
          float gN = tanhf(acc[2][r]);
          float oN = sigm(acc[3][r]);
          float ccv = fN * cc[jt][r] + iN * gN;
          cc[jt][r] = ccv;
          float hv = oN * tanhf(ccv);
          int mrow = r + 8 * lhalf;
          int col = jt * 16 + lrow;
          sHH[mrow * HD + col] = (bf16)hv;
          int node = nodebase + mrow;
          if (storeSeq) yseq[((size_t)node * KK + t) * HD + col] = (bf16)hv;
          if (t == lastv[r]) ylast[(size_t)node * 256 + d * 128 + col] = hv;
        }
      }
    }
  }
}

// f gate: f_acc[n][j] = sum_k sigmoid(Wf[n][j] + y[n,k]@fcW^T[j]) * c[n,k][j] * mask
__global__ __launch_bounds__(128)
void fc_seq_kernel(const bf16* __restrict__ yfseq, const bf16* __restrict__ ybseq,
                   const bf16* __restrict__ fcwb, const float* __restrict__ wx,
                   const int* __restrict__ ind, const float* __restrict__ ctab,
                   float* __restrict__ fac) {
  const int wave = threadIdx.x >> 5, lane = threadIdx.x & 31;
  const int lrow = lane & 15, lhalf = lane >> 4;
  const int n = blockIdx.x * 4 + wave;
  v16bf a[8];
#pragma unroll
  for (int kf = 0; kf < 4; ++kf)
    a[kf] = load_a_bf16(yfseq + ((size_t)n * KK + lrow) * HD, kf, lhalf);
#pragma unroll
  for (int kf = 0; kf < 4; ++kf)
    a[4 + kf] = load_a_bf16(ybseq + ((size_t)n * KK + lrow) * HD, kf, lhalf);
#pragma unroll
  for (int jt = 0; jt < 8; ++jt) {
    v8f acc = splat8(0.0f);
#pragma unroll
    for (int kf = 0; kf < 8; ++kf)
      acc = wmma_bf16(a[kf], load_b_bf16(fcwb, jt * 16 + lrow, 256, kf * 32, lhalf), acc);
    int j = jt * 16 + lrow;
    float wf = wx[(size_t)n * G4 + j];
    float partial = 0.0f;
#pragma unroll
    for (int r = 0; r < 8; ++r) {
      int k = r + 8 * lhalf;
      float bfv = sigm(wf + acc[r]);
      int iv = ind[n * KK + k];
      float cv = 0.0f;
      if (ctab != nullptr && iv != -1) cv = ctab[(size_t)(iv - 1) * HD + j];
      partial += bfv * cv;
    }
    partial += __shfl_xor(partial, 16, 32);
    if (lhalf == 0) fac[(size_t)n * HD + j] = partial;
  }
}

// gate pre-activation for i/u/o modules: gpre[n][j] = ylast[n][:256] @ fcW^T
__global__ __launch_bounds__(128)
void fc_last_kernel(const float* __restrict__ ylast, const bf16* __restrict__ fcwb,
                    float* __restrict__ gpre) {
  const int wave = threadIdx.x >> 5, lane = threadIdx.x & 31;
  const int lrow = lane & 15, lhalf = lane >> 4;
  const int nodebase = (blockIdx.x * 4 + wave) * 16;
  const float* arow = ylast + (size_t)(nodebase + lrow) * 256;
  v16bf a[8];
#pragma unroll
  for (int kf = 0; kf < 8; ++kf) a[kf] = load_a_f32(arow, kf, lhalf);
#pragma unroll
  for (int jt = 0; jt < 8; ++jt) {
    v8f acc = splat8(0.0f);
#pragma unroll
    for (int kf = 0; kf < 8; ++kf)
      acc = wmma_bf16(a[kf], load_b_bf16(fcwb, jt * 16 + lrow, 256, kf * 32, lhalf), acc);
#pragma unroll
    for (int r = 0; r < 8; ++r)
      gpre[(size_t)(nodebase + r + 8 * lhalf) * HD + jt * 16 + lrow] = acc[r];
  }
}

__global__ __launch_bounds__(256)
void combine_kernel(const float* __restrict__ wx, const float* __restrict__ gi,
                    const float* __restrict__ gu, const float* __restrict__ go,
                    const float* __restrict__ fac, float* __restrict__ outh,
                    float* __restrict__ outc) {
  size_t id = (size_t)blockIdx.x * 256 + threadIdx.x;  // over N*H
  size_t n = id >> 7;
  int j = (int)(id & 127);
  const float* wxr = wx + n * G4;
  float iv = sigm(gi[id] + wxr[128 + j]);
  float uv = tanhf(gu[id] + wxr[256 + j]);
  float ov = sigm(go[id] + wxr[384 + j]);
  float nc = iv * uv + fac[id];
  float nh = ov * tanhf(nc);
  outh[id] = nh;
  outc[id] = nc;
}

// ---------------- host launch ----------------
extern "C" void kernel_launch(void* const* d_in, const int* in_sizes, int n_in,
                              void* d_out, int out_size, void* d_ws, size_t ws_size,
                              hipStream_t stream) {
  (void)in_sizes; (void)n_in; (void)out_size; (void)ws_size;
  const float* x = (const float*)d_in[0];
  const int* indices = (const int*)d_in[1];
  const float* Ww = (const float*)d_in[54];
  const float* Wbias = (const float*)d_in[55];

  char* ws = (char*)d_ws;
  size_t off = 0;
  auto take = [&](size_t bytes) -> char* {
    char* p = ws + off;
    off = (off + bytes + 255) & ~(size_t)255;
    return p;
  };

  bf16 *wihb[4][2], *whhb[4][2], *fcwb[4];
  float* bsum[4][2];
  for (int m = 0; m < 4; ++m)
    for (int d = 0; d < 2; ++d) {
      wihb[m][d] = (bf16*)take(G4 * HD * sizeof(bf16));
      whhb[m][d] = (bf16*)take(G4 * HD * sizeof(bf16));
      bsum[m][d] = (float*)take(G4 * sizeof(float));
    }
  for (int m = 0; m < 4; ++m) fcwb[m] = (bf16*)take(HD * 256 * sizeof(bf16));
  bf16* wwb = (bf16*)take(G4 * DIN * sizeof(bf16));
  int* lenb = (int*)take(NN * sizeof(int));
  bf16* hfw = (bf16*)take((size_t)NN * KK * HD * sizeof(bf16));
  float* wxbuf = (float*)take((size_t)NN * G4 * sizeof(float));
  bf16* yf = (bf16*)take((size_t)NN * KK * HD * sizeof(bf16));
  bf16* yb = (bf16*)take((size_t)NN * KK * HD * sizeof(bf16));
  float* ylast = (float*)take((size_t)NN * 256 * sizeof(float));
  float* gpre[3];
  for (int i = 0; i < 3; ++i) gpre[i] = (float*)take((size_t)NN * HD * sizeof(float));
  float* fac = (float*)take((size_t)NN * HD * sizeof(float));

  // weight prep (fp32 -> bf16, bias sums)
  for (int m = 0; m < 4; ++m) {
    const int b = 2 + m * 13;
    cvt_kernel<<<256, 256, 0, stream>>>((const float*)d_in[b + 4], wihb[m][0], G4 * HD);
    cvt_kernel<<<256, 256, 0, stream>>>((const float*)d_in[b + 5], whhb[m][0], G4 * HD);
    cvt_kernel<<<256, 256, 0, stream>>>((const float*)d_in[b + 8], wihb[m][1], G4 * HD);
    cvt_kernel<<<256, 256, 0, stream>>>((const float*)d_in[b + 9], whhb[m][1], G4 * HD);
    cvt_kernel<<<128, 256, 0, stream>>>((const float*)d_in[b + 12], fcwb[m], HD * 256);
    bias_kernel<<<2, 256, 0, stream>>>((const float*)d_in[b + 6],
                                       (const float*)d_in[b + 7], bsum[m][0]);
    bias_kernel<<<2, 256, 0, stream>>>((const float*)d_in[b + 10],
                                       (const float*)d_in[b + 11], bsum[m][1]);
  }
  cvt_kernel<<<512, 256, 0, stream>>>(Ww, wwb, G4 * DIN);

  float* resh = (float*)d_out;
  float* resc = resh + (size_t)LVLS * NN * HD;
  const size_t SMEM = (size_t)2 * G4 * HD * sizeof(bf16) + 4 * 16 * HD * sizeof(bf16);

  for (int l = 0; l < LVLS; ++l) {
    const int* indl = indices + (size_t)l * NN * KK;
    const float* htab = l ? (resh + (size_t)(l - 1) * NN * HD) : nullptr;
    const float* ctab = l ? (resc + (size_t)(l - 1) * NN * HD) : nullptr;
    gather_kernel<<<(NN * KK * HD) / 256, 256, 0, stream>>>(indl, htab, hfw);
    len_kernel<<<NN / 256, 256, 0, stream>>>(indl, lenb);
    wx_kernel<<<NN / 64, 128, 0, stream>>>(x + (size_t)l * NN * DIN, wwb, Wbias, wxbuf);
    for (int m = 0; m < 4; ++m) {
      const int b = 2 + m * 13;
      bilstm_kernel<<<NN / 64, 128, SMEM, stream>>>(
          hfw, lenb, wihb[m][0], whhb[m][0], wihb[m][1], whhb[m][1],
          bsum[m][0], bsum[m][1],
          (const float*)d_in[b + 0], (const float*)d_in[b + 1],
          (const float*)d_in[b + 2], (const float*)d_in[b + 3],
          yf, yb, ylast, (m == 0) ? 1 : 0);
      if (m == 0)
        fc_seq_kernel<<<NN / 4, 128, 0, stream>>>(yf, yb, fcwb[0], wxbuf, indl, ctab, fac);
      else
        fc_last_kernel<<<NN / 64, 128, 0, stream>>>(ylast, fcwb[m], gpre[m - 1]);
    }
    combine_kernel<<<(NN * HD) / 256, 256, 0, stream>>>(
        wxbuf, gpre[0], gpre[1], gpre[2], fac,
        resh + (size_t)l * NN * HD, resc + (size_t)l * NN * HD);
  }
}
